// MultiHeadedAttention_26886495273736
// MI455X (gfx1250) — matrix operand order repaired and checked by reading, not compiled, not run on hardware
//
#include <hip/hip_runtime.h>
#include <hip/hip_bf16.h>

typedef __attribute__((ext_vector_type(16))) __bf16 v16bf;
typedef __attribute__((ext_vector_type(8)))  __bf16 v8bf;
typedef __attribute__((ext_vector_type(8)))  float  v8f;
typedef __attribute__((ext_vector_type(4)))  float  v4f;
typedef __attribute__((ext_vector_type(4)))  unsigned int u32x4;
typedef __attribute__((ext_vector_type(8)))  int    i32x8;
typedef __attribute__((ext_vector_type(4)))  int    i32x4;

#define BS        16
#define SEQ_LEN   48
#define N_JOINTS  24
#define NTOK      (SEQ_LEN * N_JOINTS)   /* 1152 */
#define NHEADS    8
#define DHEAD     32
#define DMODEL    256
#define DIN       256

// two contiguous 16B runs -> one 16x32-bf16 WMMA operand
__device__ __forceinline__ v16bf load_frag16(const __bf16* p0, const __bf16* p1) {
    v8bf lo = *(const v8bf*)p0;
    v8bf hi = *(const v8bf*)p1;
    return __builtin_shufflevector(lo, hi, 0,1,2,3,4,5,6,7,8,9,10,11,12,13,14,15);
}

// ---------------------------------------------------------------------------
// Prep 1: x (f32) -> xbf (bf16), row major.  8 elems / thread.
// ---------------------------------------------------------------------------
__global__ __launch_bounds__(256)
void cvt_x_bf16_kernel(const float* __restrict__ x, __bf16* __restrict__ xbf)
{
    const size_t base = ((size_t)blockIdx.x * 256 + threadIdx.x) * 8;
    const v4f f0 = *(const v4f*)(x + base);
    const v4f f1 = *(const v4f*)(x + base + 4);
    v8bf o;
    #pragma unroll
    for (int i = 0; i < 4; ++i) { o[i] = (__bf16)f0[i]; o[4 + i] = (__bf16)f1[i]; }
    *(v8bf*)(xbf + base) = o;
}

// ---------------------------------------------------------------------------
// Prep 2: Wt[z][n][k] = W[z][k][n] as bf16 (so B fragments are contiguous).
// ---------------------------------------------------------------------------
__global__ __launch_bounds__(256)
void transpose_w_kernel(const float* __restrict__ Wq, const float* __restrict__ Wk,
                        const float* __restrict__ Wv, __bf16* __restrict__ wt)
{
    const int c  = blockIdx.x * 256 + threadIdx.x;       // 3*256*32 chunks
    const int z  = c / (DMODEL * (DIN / 8));
    const int r  = c % (DMODEL * (DIN / 8));
    const int n  = r / (DIN / 8);
    const int k8 = (r % (DIN / 8)) * 8;
    const float* W = (z == 0) ? Wq : (z == 1) ? Wk : Wv;
    v8bf o;
    #pragma unroll
    for (int i = 0; i < 8; ++i) o[i] = (__bf16)W[(size_t)(k8 + i) * DMODEL + n];
    *(v8bf*)(wt + ((size_t)z * DMODEL + n) * DIN + k8) = o;
}

// ---------------------------------------------------------------------------
// Projection: qkv[z][b][h][tok][d] (bf16) = relu?(X @ W[z] + b[z])
// No LDS: fragments stream from the 192MB L2. 8 waves, 64(M)x256(N) per WG.
// Per K-chunk: issue all 16 B loads + 2 A loads in clauses, one wait, then
// 8 back-to-back WMMAs (XDL pipelining, no per-tile loadcnt stalls).
// ---------------------------------------------------------------------------
__global__ __launch_bounds__(256)
void proj_qkv_kernel(const __bf16* __restrict__ xbf, const __bf16* __restrict__ wt,
                     const float* __restrict__ bq, const float* __restrict__ bk,
                     const float* __restrict__ bv, __bf16* __restrict__ qkv)
{
    const int tid  = threadIdx.x;
    const int lane = tid & 31;
    const int wave = tid >> 5;
    const int mblk = blockIdx.x;             // 0..17  (64 rows)
    const int b    = blockIdx.y;
    const int z    = blockIdx.z;             // 0=Q 1=K 2=V

    const float* bias = (z == 0) ? bq : (z == 1) ? bk : bv;
    const int tok0  = mblk * 64;
    const int m0    = (wave >> 1) * 16;      // M sub-tile
    const int npart = wave & 1;              // N half
    const int kb    = (lane >> 4) * 8;       // table-order K base (unused)
    const int kb2   = (lane >> 4) * 16;      // K base for A and B (contiguous k = 16h + s)
    const int ncol  = lane & 15;

    const __bf16* Xrow = xbf + ((size_t)b * NTOK + tok0 + m0 + ncol) * DIN;
    // per-lane base of this wave's first B column run; tile j adds j*16 rows
    const __bf16* Wbase = wt + (size_t)z * DMODEL * DIN
                             + (size_t)((npart * 8) * 16 + ncol) * DIN + kb2;

    v8f acc[8];
    #pragma unroll
    for (int j = 0; j < 8; ++j)
        #pragma unroll
        for (int r = 0; r < 8; ++r) acc[j][r] = 0.0f;

    #pragma unroll 1
    for (int kc = 0; kc < DIN / 32; ++kc) {
        const int k0 = kc * 32;

        // ---- gather phase: A + all 8 B fragments, then compute phase ----
        v16bf a = load_frag16(Xrow + k0 + kb2, Xrow + k0 + kb2 + 8);
        v16bf bb[8];
        #pragma unroll
        for (int j = 0; j < 8; ++j) {
            const __bf16* wp = Wbase + (size_t)j * 16 * DIN + k0;
            bb[j] = load_frag16(wp, wp + 8);
        }
        if (kc + 1 < DIN / 32) __builtin_prefetch(Xrow + k0 + 32, 0, 3);

        #pragma unroll
        for (int j = 0; j < 8; ++j)
            acc[j] = __builtin_amdgcn_wmma_f32_16x16x32_bf16(
                         false, a, false, bb[j], (short)0, acc[j], false, false);
    }

    // epilogue: + bias (ReLU for V), bf16 scatter to qkv[z][b][h][tok][d]
    #pragma unroll
    for (int j = 0; j < 8; ++j) {
        const int col    = (npart * 8 + j) * 16 + ncol;
        const int head   = col >> 5;
        const int d      = col & 31;
        const float bc   = bias[col];
        __bf16* outp = qkv + ((((size_t)z * BS + b) * NHEADS + head) * NTOK
                               + tok0 + m0 + 8 * (lane >> 4)) * DHEAD + d;
        #pragma unroll
        for (int r = 0; r < 8; ++r) {
            float vv = acc[j][r] + bc;
            if (z == 2) vv = fmaxf(vv, 0.0f);
            outp[(size_t)r * DHEAD] = (__bf16)vv;
        }
    }
}

// ---------------------------------------------------------------------------
// Spatial attention: WG per (batch, frame); wave per head. 24x24 blocks.
// Q/K fragments stream straight from global (contiguous per lane).
// V tile staged to LDS by the Tensor Data Mover (fallback: vector loads),
// transposed once so the P@V B-fragment is contiguous.
// ---------------------------------------------------------------------------
__global__ __launch_bounds__(256)
void spatial_attn_kernel(const __bf16* __restrict__ qkv, float* __restrict__ out)
{
    __shared__ __attribute__((aligned(16))) float  sS[NHEADS][32][33]; // scores
    __shared__ __attribute__((aligned(16))) __bf16 sP[NHEADS][32][40]; // probs
    __shared__ __attribute__((aligned(16))) __bf16 sVr[NHEADS][32][32];// V rows
    __shared__ __attribute__((aligned(16))) __bf16 sVt[NHEADS][32][40];// V^T [d][key]

    const int lane = threadIdx.x & 31;
    const int wave = threadIdx.x >> 5;       // head
    const int b    = blockIdx.x / SEQ_LEN;
    const int f    = blockIdx.x % SEQ_LEN;
    const int tok0 = f * N_JOINTS;
    const int kb   = (lane >> 4) * 8;
    const int kb2  = (lane >> 4) * 16;
    const int ncol = lane & 15;

    const size_t mat  = (size_t)BS * NHEADS * NTOK * DHEAD;
    const size_t base = (((size_t)b * NHEADS + wave) * NTOK + tok0) * DHEAD;
    const __bf16* Q = qkv + base;
    const __bf16* K = qkv + mat + base;
    const __bf16* V = qkv + 2 * mat + base;

    // ---- stage V tile (24 x 32 bf16) into sVr via the Tensor Data Mover ----
#if defined(__gfx1250__) && __has_builtin(__builtin_amdgcn_tensor_load_to_lds) && \
    __has_builtin(__builtin_amdgcn_s_wait_tensorcnt)
    {
        const unsigned lds_addr     = (unsigned)(unsigned long long)(const void*)&sVr[wave][0][0];
        const unsigned long long ga = (unsigned long long)(const void*)V;
        u32x4 g0 = {0, 0, 0, 0};
        i32x8 g1 = {0, 0, 0, 0, 0, 0, 0, 0};
        i32x4 g2 = {0, 0, 0, 0};
        i32x4 g3 = {0, 0, 0, 0};
        i32x8 g4 = {0, 0, 0, 0, 0, 0, 0, 0};
        g0[0] = 1u;                                              // count = 1
        g0[1] = lds_addr;                                        // lds_addr
        g0[2] = (unsigned)(ga & 0xFFFFFFFFull);                  // global_addr lo
        g0[3] = (unsigned)((ga >> 32) & 0x01FFFFFFull) | (2u << 30); // hi | type=2
        g1[0] = (int)(1u << 16);                 // data_size = 1 (2 bytes/elem)
        g1[1] = (int)((unsigned)DHEAD << 16);    // tensor_dim0 = 32   (bits 79:48)
        g1[2] = (int)((unsigned)N_JOINTS << 16); // tensor_dim1 = 24   (bits 111:80)
        g1[3] = (int)((unsigned)DHEAD << 16);    // tile_dim0 = 32     (bits 127:112)
        g1[4] = (int)N_JOINTS;                   // tile_dim1 = 24, tile_dim2 = 0
        g1[5] = (int)DHEAD;                      // tensor_dim0_stride = 32
        __builtin_amdgcn_tensor_load_to_lds(g0, g1, g2, g3, g4, 0);
        __builtin_amdgcn_s_wait_tensorcnt((short)0);
    }
#else
    if (lane < N_JOINTS) {
        const __bf16* vp = V + (size_t)lane * DHEAD;
        *(v8bf*)&sVr[wave][lane][0]  = *(const v8bf*)(vp);
        *(v8bf*)&sVr[wave][lane][8]  = *(const v8bf*)(vp + 8);
        *(v8bf*)&sVr[wave][lane][16] = *(const v8bf*)(vp + 16);
        *(v8bf*)&sVr[wave][lane][24] = *(const v8bf*)(vp + 24);
    }
#endif
    __syncthreads();

    // transpose: sVt[d][key]  (lane = d); pad keys 24..31 with zeros
    #pragma unroll
    for (int key = 0; key < 32; ++key)
        sVt[wave][lane][key] = (key < N_JOINTS) ? sVr[wave][key][lane] : (__bf16)0.0f;

    // ---- scores = Q @ K^T (d_head = 32 -> one WMMA per 16x16 tile) ----
    v8f sc[2][2];
    #pragma unroll
    for (int mt = 0; mt < 2; ++mt)
        #pragma unroll
        for (int nt = 0; nt < 2; ++nt)
            #pragma unroll
            for (int r = 0; r < 8; ++r) sc[mt][nt][r] = 0.0f;

    #pragma unroll
    for (int mt = 0; mt < 2; ++mt) {
        const __bf16* qp = Q + (size_t)(mt * 16 + ncol) * DHEAD;
        v16bf a = load_frag16(qp + kb2, qp + kb2 + 8);
        #pragma unroll
        for (int nt = 0; nt < 2; ++nt) {
            const __bf16* kp = K + (size_t)(nt * 16 + ncol) * DHEAD + kb2;
            v16bf bb = load_frag16(kp, kp + 8);              // K^T fragment
            sc[mt][nt] = __builtin_amdgcn_wmma_f32_16x16x32_bf16(
                             false, a, false, bb, (short)0, sc[mt][nt], false, false);
        }
    }

    // scale (mask/sqrt(d)) and dump 32x32 scores to LDS
    const float scale = 0.17677669529663687f;   // 1/sqrt(32)
    #pragma unroll
    for (int mt = 0; mt < 2; ++mt)
        #pragma unroll
        for (int nt = 0; nt < 2; ++nt)
            #pragma unroll
            for (int r = 0; r < 8; ++r) {
                const int m = mt * 16 + r + 8 * (lane >> 4);
                const int n = nt * 16 + ncol;
                sS[wave][m][n] = sc[mt][nt][r] * scale;
            }
    __syncthreads();

    // ---- row softmax over 24 valid keys (one lane per row); pads -> 0 ----
    {
        const int rr = lane;
        float inv = 0.0f;
        if (rr < N_JOINTS) {
            float mx = -3.402823466e38f;
            #pragma unroll
            for (int n = 0; n < N_JOINTS; ++n) mx = fmaxf(mx, sS[wave][rr][n]);
            float sum = 0.0f;
            #pragma unroll
            for (int n = 0; n < N_JOINTS; ++n) {
                const float ev = __expf(sS[wave][rr][n] - mx);
                sS[wave][rr][n] = ev;
                sum += ev;
            }
            inv = 1.0f / (sum + 1e-6f);
        }
        #pragma unroll
        for (int n = 0; n < 32; ++n) {
            const float p = (rr < N_JOINTS && n < N_JOINTS) ? sS[wave][rr][n] * inv
                                                            : 0.0f;
            sP[wave][rr][n] = (__bf16)p;
        }
    }
    __syncthreads();

    // ---- out = P @ V ----
    #pragma unroll
    for (int mt = 0; mt < 2; ++mt) {
        const __bf16* pp = &sP[wave][mt * 16 + ncol][0];
        v16bf a = load_frag16(pp + kb2, pp + kb2 + 8);
        #pragma unroll
        for (int nt = 0; nt < 2; ++nt) {
            const int col = nt * 16 + ncol;                  // d index
            v16bf bb = load_frag16(&sVt[wave][col][kb2], &sVt[wave][col][kb2 + 8]);
            v8f o;
            #pragma unroll
            for (int r = 0; r < 8; ++r) o[r] = 0.0f;
            o = __builtin_amdgcn_wmma_f32_16x16x32_bf16(
                    false, a, false, bb, (short)0, o, false, false);
            #pragma unroll
            for (int r = 0; r < 8; ++r) {
                const int q = mt * 16 + r + 8 * (lane >> 4);
                if (q < N_JOINTS)
                    out[((size_t)b * NTOK + tok0 + q) * DMODEL + wave * DHEAD + col] = o[r];
            }
        }
    }
}

extern "C" void kernel_launch(void* const* d_in, const int* in_sizes, int n_in,
                              void* d_out, int out_size, void* d_ws, size_t ws_size,
                              hipStream_t stream) {
    (void)in_sizes; (void)n_in; (void)out_size; (void)ws_size;
    const float* x  = (const float*)d_in[0];
    const float* Wq = (const float*)d_in[1];
    const float* bq = (const float*)d_in[2];
    const float* Wk = (const float*)d_in[3];
    const float* bk = (const float*)d_in[4];
    const float* Wv = (const float*)d_in[5];
    const float* bv = (const float*)d_in[6];
    float* out = (float*)d_out;

    // bf16 workspace layout: [qkv | xbf | wt]  (qkv first so 8-row frame
    // over-reads from padded fragments land inside the workspace)
    __bf16* wsb = (__bf16*)d_ws;
    __bf16* qkv = wsb;                                        // 3*16*8*1152*32
    __bf16* xbf = wsb + (size_t)3 * BS * NHEADS * NTOK * DHEAD;
    __bf16* wt  = xbf + (size_t)BS * NTOK * DIN;              // 3*256*256

    cvt_x_bf16_kernel<<<dim3((BS * NTOK * DIN) / (256 * 8)), 256, 0, stream>>>(x, xbf);
    transpose_w_kernel<<<dim3((3 * DMODEL * DIN / 8) / 256), 256, 0, stream>>>(Wq, Wk, Wv, wt);

    dim3 gridP(NTOK / 64, BS, 3);
    proj_qkv_kernel<<<gridP, 256, 0, stream>>>(xbf, wt, bq, bk, bv, qkv);

    spatial_attn_kernel<<<dim3(BS * SEQ_LEN), 256, 0, stream>>>(qkv, out);
}
